// PatchedDeepseekAttention_75058848465333
// MI455X (gfx1250) — compile-verified, matrix operand order checked by reading, not run on hardware
//
#include <hip/hip_runtime.h>

// ---------------- types ----------------
typedef __attribute__((ext_vector_type(16))) __bf16 v16bf;
typedef __attribute__((ext_vector_type(8)))  __bf16 v8bf;
typedef __attribute__((ext_vector_type(8)))  float  v8f;
typedef __attribute__((ext_vector_type(4)))  unsigned int u32x4;
typedef __attribute__((ext_vector_type(8)))  int i32x8;
typedef __attribute__((ext_vector_type(4)))  int i32x4;

#define T_TOK   4096
#define HID     4096
#define NHEADS  32
#define HD      128
#define QLEN    512
#define KVLEN   1024
#define BATCH   8
#define HISTLEN 512

// softmax scale folded into log2 domain: (1/sqrt(128)) * log2(e)
#define SCALE_L2E (0.08838834764831845f * 1.4426950408889634f)

__device__ __forceinline__ unsigned short f2bf(float x) {
  unsigned u = __builtin_bit_cast(unsigned, x);
  u += 0x7FFFu + ((u >> 16) & 1u);          // round-to-nearest-even
  return (unsigned short)(u >> 16);
}
__device__ __forceinline__ void stbf(__bf16* p, float x) {
  *(unsigned short*)p = f2bf(x);
}

// 16x32 bf16 WMMA fragment from a row-major global operand.
// Per ISA 7.12.2: lanes 0-15 hold K=0..7 / 16..23, lanes 16-31 hold K=8..15 / 24..31.
__device__ __forceinline__ v16bf load_frag(const __bf16* __restrict__ base,
                                           int rowStride, int rowBase, int k0,
                                           int lane) {
  int r  = lane & 15;
  int kb = k0 + ((lane >> 4) << 3);
  const __bf16* p = base + (size_t)(rowBase + r) * rowStride + kb;
  union { v16bf v; v8bf h[2]; } u;
  u.h[0] = *(const v8bf*)(p);
  u.h[1] = *(const v8bf*)(p + 16);
  return u.v;
}

// 16x32 bf16 WMMA fragment from an LDS tile with row stride 32 elements.
__device__ __forceinline__ v16bf lds_frag(const __bf16* tile, int rowBase, int lane) {
  int r  = lane & 15;
  int kb = (lane >> 4) << 3;
  const __bf16* p = tile + (rowBase + r) * 32 + kb;
  union { v16bf v; v8bf h[2]; } u;
  u.h[0] = *(const v8bf*)(p);
  u.h[1] = *(const v8bf*)(p + 16);
  return u.v;
}

// ---------------- Tensor Data Mover: 2-D bf16 tile -> LDS ----------------
// D# per ISA 08_async_tensor §8.3/§8.4: group0 = {flags, lds_addr, global_addr, type=2},
// group1 = {data_size=2B, tensor_dim0/1, tile_dim0/1, tensor_dim0_stride}.
// This toolchain exposes the 6-arg builtin: (g0, g1, g2, g3, g4, cpol).
__device__ __forceinline__ void tdm_load_tile_bf16(const __bf16* gsrc,
                                                   unsigned ldsOff,
                                                   int dim0, int dim1,
                                                   int tile0, int tile1,
                                                   int stride0) {
  unsigned long long ga = (unsigned long long)(const void*)gsrc;
  u32x4 g0;
  g0.x = 1u;                                                   // count=1, user mode
  g0.y = ldsOff;                                               // lds_addr (bytes)
  g0.z = (unsigned)ga;                                         // global_addr[31:0]
  g0.w = (unsigned)((ga >> 32) & 0x01FFFFFFull) | (2u << 30);  // [56:32] | type=2
  i32x8 g1;
  g1[0] = (int)(1u << 16);                                     // data_size=1 -> 2 bytes
  g1[1] = (int)(((unsigned)dim0 & 0xFFFFu) << 16);             // tensor_dim0 lo
  g1[2] = (int)((((unsigned)dim0 >> 16) & 0xFFFFu)
              | (((unsigned)dim1 & 0xFFFFu) << 16));           // dim0 hi | dim1 lo
  g1[3] = (int)((((unsigned)dim1 >> 16) & 0xFFFFu)
              | ((unsigned)tile0 << 16));                      // dim1 hi | tile_dim0
  g1[4] = (int)((unsigned)tile1 & 0xFFFFu);                    // tile_dim1 (tile_dim2=0)
  g1[5] = (int)stride0;                                        // tensor_dim0_stride lo
  g1[6] = 0;                                                   // stride hi / dim1_stride
  g1[7] = 0;
  i32x4 gz4 = (i32x4){0, 0, 0, 0};
  i32x8 gz8 = (i32x8){0, 0, 0, 0, 0, 0, 0, 0};
  __builtin_amdgcn_tensor_load_to_lds(g0, g1, gz4, gz4, gz8, 0);
}

__device__ __forceinline__ unsigned lds_off(const void* p) {
  return (unsigned)(unsigned long long)p;   // low 32 bits of generic LDS addr = LDS offset
}

// ---------------- elementwise prep kernels ----------------
__global__ __launch_bounds__(256) void conv_hidden_kernel(const float* __restrict__ H,
                                                          __bf16* __restrict__ Hb) {
  int tid = blockIdx.x * 256 + threadIdx.x;
  if (tid < T_TOK * HID) stbf(&Hb[tid], H[tid]);
}

__global__ __launch_bounds__(256) void transpose_w_kernel(const float* __restrict__ W,
                                                          __bf16* __restrict__ WT) {
  int tid = blockIdx.x * 256 + threadIdx.x;     // over K*N = 16M
  if (tid >= HID * HID) return;
  int n = tid & (HID - 1);
  int k = tid >> 12;
  stbf(&WT[(size_t)n * HID + k], W[(size_t)k * HID + n]);
}

// ---------------- TDM + WMMA GEMM: C[M,N] = A[M,K](bf16) x Bt[N,K]^T ----------------
// 128x128 block tile, 4 waves each computing 64x64. Double-buffered LDS tiles
// filled by the Tensor Data Mover; fragments read from LDS (ds_load_b128).
__global__ __launch_bounds__(128) void gemm_bf16_wmma(const __bf16* __restrict__ A,
                                                      const __bf16* __restrict__ Bt,
                                                      float* __restrict__ C,
                                                      int M, int N, int K) {
  __shared__ __align__(16) __bf16 As[2][128 * 32];   // 8KB x2
  __shared__ __align__(16) __bf16 Bs[2][128 * 32];   // 8KB x2

  int lane = threadIdx.x & 31;
  int wave = threadIdx.x >> 5;                  // 4 waves (wave32)
  int wm = wave >> 1, wn = wave & 1;
  int mBlk = blockIdx.y * 128;
  int nBlk = blockIdx.x * 128;

  v8f acc[4][4];
#pragma unroll
  for (int i = 0; i < 4; ++i)
#pragma unroll
    for (int j = 0; j < 4; ++j) acc[i][j] = (v8f){};

  if (wave == 0) {   // TDM issues once per wave; EXEC ignored
    tdm_load_tile_bf16(A  + (size_t)mBlk * K, lds_off(&As[0][0]), K, M, 32, 128, K);
    tdm_load_tile_bf16(Bt + (size_t)nBlk * K, lds_off(&Bs[0][0]), K, N, 32, 128, K);
  }

  int buf = 0;
  for (int k0 = 0; k0 < K; k0 += 32, buf ^= 1) {
    if (wave == 0) __builtin_amdgcn_s_wait_tensorcnt(0);
    __syncthreads();                           // publish tile / retire reads of buf^1
    if (wave == 0 && k0 + 32 < K) {
      tdm_load_tile_bf16(A  + (size_t)mBlk * K + (k0 + 32),
                         lds_off(&As[buf ^ 1][0]), K, M, 32, 128, K);
      tdm_load_tile_bf16(Bt + (size_t)nBlk * K + (k0 + 32),
                         lds_off(&Bs[buf ^ 1][0]), K, N, 32, 128, K);
    }
    v16bf af[4], bfr[4];
#pragma unroll
    for (int i = 0; i < 4; ++i) af[i]  = lds_frag(&As[buf][0], wm * 64 + i * 16, lane);
#pragma unroll
    for (int j = 0; j < 4; ++j) bfr[j] = lds_frag(&Bs[buf][0], wn * 64 + j * 16, lane);
#pragma unroll
    for (int i = 0; i < 4; ++i)
#pragma unroll
      for (int j = 0; j < 4; ++j)
        acc[i][j] = __builtin_amdgcn_wmma_f32_16x16x32_bf16(false, af[i], false, bfr[j],
                                                            (short)0, acc[i][j], false, false);
  }

  int half = lane >> 4, col0 = lane & 15;
#pragma unroll
  for (int i = 0; i < 4; ++i)
#pragma unroll
    for (int j = 0; j < 4; ++j)
#pragma unroll
      for (int r = 0; r < 8; ++r) {
        int row = mBlk + wm * 64 + i * 16 + r + 8 * half;
        int col = nBlk + wn * 64 + j * 16 + col0;
        C[(size_t)row * N + col] = acc[i][j][r];
      }
}

// ---------------- RoPE + repack into attention layouts ----------------
__global__ __launch_bounds__(256) void rope_pack_kernel(const float* __restrict__ q,
                                                        const float* __restrict__ k,
                                                        const float* __restrict__ v,
                                                        __bf16* __restrict__ Qb,
                                                        __bf16* __restrict__ Kb,
                                                        __bf16* __restrict__ Vt) {
  int tid = blockIdx.x * 256 + threadIdx.x;     // T * NH * 64
  if (tid >= T_TOK * NHEADS * 64) return;
  int i  = tid & 63;
  int h  = (tid >> 6) & 31;
  int t  = tid >> 11;
  int b  = t >> 9;
  int qr = t & 511;
  int pos = HISTLEN + qr;

  float inv = exp2f(-((float)(2 * i) / 128.0f) * 13.287712379549449f); // 10000^(-2i/128)
  float ang = (float)pos * inv;
  float c, s;
  __sincosf(ang, &s, &c);
  size_t base = (size_t)t * HID + h * HD + i;
  float q1 = q[base], q2 = q[base + 64];
  float k1 = k[base], k2 = k[base + 64];
  float v1 = v[base], v2 = v[base + 64];
  float qo1 = q1 * c - q2 * s, qo2 = q2 * c + q1 * s;
  float ko1 = k1 * c - k2 * s, ko2 = k2 * c + k1 * s;

  size_t bh = (size_t)(b * NHEADS + h);
  __bf16* Qp = Qb + bh * QLEN * HD + (size_t)qr * HD;
  stbf(Qp + i, qo1);
  stbf(Qp + 64 + i, qo2);
  __bf16* Kp = Kb + bh * KVLEN * HD + (size_t)(HISTLEN + qr) * HD;
  stbf(Kp + i, ko1);
  stbf(Kp + 64 + i, ko2);
  __bf16* Vp = Vt + bh * HD * KVLEN;
  stbf(Vp + (size_t)i * KVLEN + HISTLEN + qr, v1);
  stbf(Vp + (size_t)(64 + i) * KVLEN + HISTLEN + qr, v2);
}

// ---------------- history K/V gather via block table ----------------
__global__ __launch_bounds__(256) void gather_hist_kernel(const float* __restrict__ kc,
                                                          const float* __restrict__ vc,
                                                          const int* __restrict__ blk_tab,
                                                          __bf16* __restrict__ Kb,
                                                          __bf16* __restrict__ Vt) {
  int tid = blockIdx.x * 256 + threadIdx.x;     // B*NH*512*128 = 16M
  if (tid >= BATCH * NHEADS * HISTLEN * HD) return;
  int d  = tid & 127;
  int kv = (tid >> 7) & 511;
  int h  = (tid >> 16) & 31;
  int b  = tid >> 21;
  int blk  = blk_tab[b * 16 + (kv >> 6)];
  int slot = kv & 63;
  size_t src = (((size_t)blk * 64 + slot) * NHEADS + h) * HD + d;
  size_t bh  = (size_t)(b * NHEADS + h);
  stbf(Kb + bh * KVLEN * HD + (size_t)kv * HD + d, kc[src]);
  stbf(Vt + bh * HD * KVLEN + (size_t)d * KVLEN + kv, vc[src]);
}

// ---------------- flash attention (online softmax, WMMA) ----------------
__global__ __launch_bounds__(128) void attn_fwd_kernel(const __bf16* __restrict__ Qb,
                                                       const __bf16* __restrict__ Kb,
                                                       const __bf16* __restrict__ Vt,
                                                       __bf16* __restrict__ Oa) {
  __shared__ __align__(16) unsigned short lds_p[4][16 * 32];   // per-wave P staging

  int lane = threadIdx.x & 31;
  int wave = threadIdx.x >> 5;
  int bh = blockIdx.x;                    // b*32 + h
  int b  = bh >> 5;
  int h  = bh & 31;
  int qbase = blockIdx.y * 64 + wave * 16;

  const __bf16* Q  = Qb + (size_t)bh * QLEN * HD;
  const __bf16* Kp = Kb + (size_t)bh * KVLEN * HD;
  const __bf16* Vp = Vt + (size_t)bh * HD * KVLEN;

  v16bf qf[4];
#pragma unroll
  for (int d = 0; d < 4; ++d) qf[d] = load_frag(Q, HD, qbase, d * 32, lane);

  v8f o[8];
#pragma unroll
  for (int dt = 0; dt < 8; ++dt) o[dt] = (v8f){};
  float m[8], l[8];
#pragma unroll
  for (int r = 0; r < 8; ++r) { m[r] = -1e30f; l[r] = 0.0f; }

  int half = lane >> 4, col0 = lane & 15;
  int qpos_min = HISTLEN + qbase;
  int kv_end = (qpos_min + 16 + 31) & ~31;

  for (int kv0 = 0; kv0 < kv_end; kv0 += 32) {
    v8f s0 = (v8f){}, s1 = (v8f){};
#pragma unroll
    for (int d = 0; d < 4; ++d) {
      v16bf kf0 = load_frag(Kp, HD, kv0,      d * 32, lane);
      v16bf kf1 = load_frag(Kp, HD, kv0 + 16, d * 32, lane);
      s0 = __builtin_amdgcn_wmma_f32_16x16x32_bf16(false, qf[d], false, kf0,
                                                   (short)0, s0, false, false);
      s1 = __builtin_amdgcn_wmma_f32_16x16x32_bf16(false, qf[d], false, kf1,
                                                   (short)0, s1, false, false);
    }

#pragma unroll
    for (int r = 0; r < 8; ++r) {
      int qp = qpos_min + r + 8 * half;
      s0[r] = ((kv0 + col0)      <= qp) ? s0[r] * SCALE_L2E : -1e30f;
      s1[r] = ((kv0 + 16 + col0) <= qp) ? s1[r] * SCALE_L2E : -1e30f;
    }

#pragma unroll
    for (int r = 0; r < 8; ++r) {
      float vmax = fmaxf(s0[r], s1[r]);
      vmax = fmaxf(vmax, __shfl_xor(vmax, 1, 16));
      vmax = fmaxf(vmax, __shfl_xor(vmax, 2, 16));
      vmax = fmaxf(vmax, __shfl_xor(vmax, 4, 16));
      vmax = fmaxf(vmax, __shfl_xor(vmax, 8, 16));
      float mn = fmaxf(m[r], vmax);
      float alpha = exp2f(m[r] - mn);
      m[r] = mn;
      float p0 = exp2f(s0[r] - mn);
      float p1 = exp2f(s1[r] - mn);
      s0[r] = p0;
      s1[r] = p1;
      float rs = p0 + p1;
      rs += __shfl_xor(rs, 1, 16);
      rs += __shfl_xor(rs, 2, 16);
      rs += __shfl_xor(rs, 4, 16);
      rs += __shfl_xor(rs, 8, 16);
      l[r] = l[r] * alpha + rs;
#pragma unroll
      for (int dt = 0; dt < 8; ++dt) o[dt][r] *= alpha;
    }

    // stage P through LDS: C-layout -> A-layout (16x32 bf16), wave-private region
    unsigned short* my = lds_p[wave];
#pragma unroll
    for (int r = 0; r < 8; ++r) {
      int row = r + 8 * half;
      my[row * 32 + col0]      = f2bf(s0[r]);
      my[row * 32 + 16 + col0] = f2bf(s1[r]);
    }
    union { v16bf v; v8bf hh[2]; } pu;
    {
      const __bf16* lp = (const __bf16*)&my[(lane & 15) * 32 + (half << 3)];
      pu.hh[0] = *(const v8bf*)(lp);
      pu.hh[1] = *(const v8bf*)(lp + 16);
    }

#pragma unroll
    for (int dt = 0; dt < 8; ++dt) {
      v16bf vfrag = load_frag(Vp, KVLEN, dt * 16, kv0, lane);
      o[dt] = __builtin_amdgcn_wmma_f32_16x16x32_bf16(false, pu.v, false, vfrag,
                                                      (short)0, o[dt], false, false);
    }
  }

  size_t trow = (size_t)(b * QLEN + qbase);
#pragma unroll
  for (int dt = 0; dt < 8; ++dt)
#pragma unroll
    for (int r = 0; r < 8; ++r) {
      int row = r + 8 * half;
      float val = o[dt][r] / l[r];
      size_t idx = (trow + row) * (size_t)HID + h * HD + dt * 16 + col0;
      stbf(&Oa[idx], val);
    }
}

// ---------------- launch ----------------
extern "C" void kernel_launch(void* const* d_in, const int* in_sizes, int n_in,
                              void* d_out, int out_size, void* d_ws, size_t ws_size,
                              hipStream_t stream) {
  (void)in_sizes; (void)n_in; (void)out_size; (void)ws_size;
  const float* hidden = (const float*)d_in[0];
  const float* Wq = (const float*)d_in[1];
  const float* Wk = (const float*)d_in[2];
  const float* Wv = (const float*)d_in[3];
  const float* Wo = (const float*)d_in[4];
  const float* kc = (const float*)d_in[5];
  const float* vc = (const float*)d_in[6];
  const int*  blk = (const int*)d_in[7];

  char* w = (char*)d_ws;
  const size_t MB = 1024ull * 1024ull;
  __bf16* WqT = (__bf16*)(w + 0 * MB);     // 32MB each
  __bf16* WkT = (__bf16*)(w + 32 * MB);
  __bf16* WvT = (__bf16*)(w + 64 * MB);
  __bf16* WoT = (__bf16*)(w + 96 * MB);
  __bf16* Hb  = (__bf16*)(w + 128 * MB);   // 32MB
  float*  qf  = (float*)(w + 160 * MB);    // 64MB each
  float*  kf  = (float*)(w + 224 * MB);
  float*  vf  = (float*)(w + 288 * MB);
  __bf16* Qb  = (__bf16*)(w + 352 * MB);   // 32MB
  __bf16* Kb  = (__bf16*)(w + 384 * MB);   // 64MB
  __bf16* Vt  = (__bf16*)(w + 448 * MB);   // 64MB
  __bf16* Ab  = (__bf16*)(w + 512 * MB);   // 32MB

  int n16 = HID * HID;
  conv_hidden_kernel<<<n16 / 256, 256, 0, stream>>>(hidden, Hb);
  transpose_w_kernel<<<n16 / 256, 256, 0, stream>>>(Wq, WqT);
  transpose_w_kernel<<<n16 / 256, 256, 0, stream>>>(Wk, WkT);
  transpose_w_kernel<<<n16 / 256, 256, 0, stream>>>(Wv, WvT);
  transpose_w_kernel<<<n16 / 256, 256, 0, stream>>>(Wo, WoT);

  dim3 gg(HID / 128, T_TOK / 128);
  gemm_bf16_wmma<<<gg, 128, 0, stream>>>(Hb, WqT, qf, T_TOK, HID, HID);
  gemm_bf16_wmma<<<gg, 128, 0, stream>>>(Hb, WkT, kf, T_TOK, HID, HID);
  gemm_bf16_wmma<<<gg, 128, 0, stream>>>(Hb, WvT, vf, T_TOK, HID, HID);

  rope_pack_kernel<<<(T_TOK * NHEADS * 64) / 256, 256, 0, stream>>>(qf, kf, vf, Qb, Kb, Vt);
  gather_hist_kernel<<<(BATCH * NHEADS * HISTLEN * HD) / 256, 256, 0, stream>>>(kc, vc, blk, Kb, Vt);

  attn_fwd_kernel<<<dim3(BATCH * NHEADS, QLEN / 64), 128, 0, stream>>>(Qb, Kb, Vt, Ab);

  gemm_bf16_wmma<<<gg, 128, 0, stream>>>(Ab, WoT, (float*)d_out, T_TOK, HID, HID);
}
